// HybridMarginLoss_32753420599900
// MI455X (gfx1250) — matrix operand order repaired and checked by reading, not compiled, not run on hardware
//
#include <hip/hip_runtime.h>
#include <hip/hip_bf16.h>

// ---------------------------------------------------------------------------
// HybridMarginLoss for MI455X (gfx1250, wave32, WMMA bf16 16x16x32)
//   loss = arcface(subcenter) + 0.5*batch_hard_triplet + 0.01*center
// Strategy: hi/lo bf16 split GEMMs on WMMA, fused group-max + online softmax,
// double-buffered async (ASYNCcnt) LDS staging of B, deterministic reductions.
// ---------------------------------------------------------------------------

namespace {
constexpr int   kN        = 2048;
constexpr int   kD        = 512;
constexpr int   kC        = 10000;
constexpr int   kCK       = 30000;   // C * K_SUB
constexpr float kScale    = 30.0f;
constexpr float kMargin   = 0.4f;
constexpr float kTripM    = 0.3f;
constexpr float kCenterW  = 0.01f;
constexpr int   kArcChunks = 25;     // CK split: 25 chunks x 25 strips x 48 cols
constexpr int   kSimChunks = 16;     // N cols split: 16 chunks x 128 cols
}

typedef __attribute__((ext_vector_type(16))) __bf16       v16bf;
typedef __attribute__((ext_vector_type(8)))  float        v8f;
typedef __attribute__((ext_vector_type(4)))  unsigned int u32x4;

union FragU { v16bf v; u32x4 q[2]; };

__device__ __forceinline__ unsigned short f2bf(float x) {
  __bf16 b = (__bf16)x; return __builtin_bit_cast(unsigned short, b);
}
__device__ __forceinline__ float bf2f(unsigned short u) {
  __bf16 b = __builtin_bit_cast(__bf16, u); return (float)b;
}

__device__ __forceinline__ v8f wmma_bf16(v16bf a, v16bf b, v8f c) {
  // D = A(16x32 bf16) * B(32x16 bf16) + C(16x16 f32)
  return __builtin_amdgcn_wmma_f32_16x16x32_bf16(false, a, false, b,
                                                 (short)0, c, false, false);
}

// CDNA5 async DMA: global memory -> LDS, 16B per lane, tracked by ASYNCcnt.
__device__ __forceinline__ void async_g2l_b128(unsigned lds_byte_addr,
                                               const void* gaddr) {
  asm volatile("global_load_async_to_lds_b128 %0, %1, off"
               :: "v"(lds_byte_addr), "v"(gaddr)
               : "memory");
}
__device__ __forceinline__ void wait_asynccnt0() {
  asm volatile("s_wait_asynccnt 0x0" ::: "memory");
}
// Low 32 bits of a generic pointer to __shared__ == wave-relative LDS address.
__device__ __forceinline__ unsigned lds_addr32(const void* p) {
  return (unsigned)(unsigned long long)p;
}

// A fragment: lane holds row M=lane&15; K = k0 + (lane>=16?16:0) covering
// {+0..15}. Two 16B chunks.
__device__ __forceinline__ v16bf ld_a_frag(const unsigned short* row, int k0, int akoff) {
  FragU f;
  const unsigned short* p = row + k0 + akoff;
  f.q[0] = *(const u32x4*)p;
  f.q[1] = *(const u32x4*)(p + 8);
  return f.v;
}

// B fragment from contiguous-K storage: lane holds col N=lane&15,
// K = k0 + (lane>=16?16:0) + 0..15. One 32B contiguous chunk.
__device__ __forceinline__ v16bf ld_b_frag(const unsigned short* p) {
  FragU f;
  f.q[0] = *(const u32x4*)p;
  f.q[1] = *(const u32x4*)(p + 8);
  return f.v;
}

// ---------------------------------------------------------------------------
// 1) Normalize embeddings; emit f32 (for center loss) + bf16 hi/lo split.
// ---------------------------------------------------------------------------
__global__ void k_eprep(const float* __restrict__ emb,
                        float* __restrict__ e_n,
                        unsigned short* __restrict__ e_hi,
                        unsigned short* __restrict__ e_lo) {
  const int row = blockIdx.x;
  const int t   = threadIdx.x;
  __shared__ float red[256];
  const float v0 = emb[(size_t)row * kD + t];
  const float v1 = emb[(size_t)row * kD + t + 256];
  red[t] = v0 * v0 + v1 * v1;
  __syncthreads();
  for (int o = 128; o > 0; o >>= 1) {
    if (t < o) red[t] += red[t + o];
    __syncthreads();
  }
  const float inv = 1.0f / fmaxf(sqrtf(red[0]), 1e-12f);
  const float a = v0 * inv, b = v1 * inv;
  const size_t i0 = (size_t)row * kD + t;
  const size_t i1 = i0 + 256;
  e_n[i0] = a; e_n[i1] = b;
  unsigned short h0 = f2bf(a), h1 = f2bf(b);
  e_hi[i0] = h0; e_hi[i1] = h1;
  e_lo[i0] = f2bf(a - bf2f(h0));
  e_lo[i1] = f2bf(b - bf2f(h1));
}

// ---------------------------------------------------------------------------
// 2) W column inverse norms + transpose to [CK, D] bf16 hi/lo.
// ---------------------------------------------------------------------------
__global__ void k_wprep(const float* __restrict__ W,
                        unsigned short* __restrict__ wt_hi,
                        unsigned short* __restrict__ wt_lo,
                        float* __restrict__ wninv) {
  const int j = blockIdx.x * blockDim.x + threadIdx.x;
  if (j >= kCK) return;
  unsigned short* oh = wt_hi + (size_t)j * kD;
  unsigned short* ol = wt_lo + (size_t)j * kD;
  union { unsigned short s[8]; u32x4 q; } bh, bl;
  float acc = 0.0f;
  for (int d = 0; d < kD; ++d) {
    const float v = W[(size_t)d * kCK + j];
    acc += v * v;
    const unsigned short h = f2bf(v);
    bh.s[d & 7] = h;
    bl.s[d & 7] = f2bf(v - bf2f(h));
    if ((d & 7) == 7) {
      *(u32x4*)(oh + d - 7) = bh.q;
      *(u32x4*)(ol + d - 7) = bl.q;
    }
  }
  wninv[j] = 1.0f / fmaxf(sqrtf(acc), 1e-12f);
}

// ---------------------------------------------------------------------------
// 3) Fused ArcFace GEMM with async double-buffered B staging.
// Grid: (16 M-blocks of 128 rows) x (25 CK-chunks of 25 strips x 48 cols).
// Block: 256 threads = 8 waves; wave w owns rows [mblk*128+w*16, +16).
// Stage st = local_strip*4 + kb; buffer = st&1; prefetch st+1 while computing.
// ---------------------------------------------------------------------------
__global__ __launch_bounds__(256, 1) void k_arc_gemm(
    const unsigned short* __restrict__ e_hi,
    const unsigned short* __restrict__ e_lo,
    const unsigned short* __restrict__ wt_hi,
    const unsigned short* __restrict__ wt_lo,
    const float* __restrict__ wninv,
    const int* __restrict__ labels,
    float* __restrict__ out_m, float* __restrict__ out_s,
    float* __restrict__ out_cl) {
  // [buf][hi/lo][48 cols x 128 K (+8 pad)]
  __shared__ __align__(16) unsigned short lsB[2][2][48 * 136];
  __shared__ float lsC[8][16][64];  // per-wave: 0..47 = cos, 48..63 = class max

  const int mblk  = blockIdx.x;   // 0..15
  const int chunk = blockIdx.y;   // 0..24
  const int t     = threadIdx.x;
  const int wid   = t >> 5;
  const int lane  = t & 31;
  const int l16   = lane & 15;
  const int hi8   = (lane & 16) ? 8 : 0;
  const int akoff = (lane & 16) ? 16 : 0;

  const int rowbase = mblk * 128 + wid * 16;
  const unsigned short* arow_h = e_hi + (size_t)(rowbase + l16) * kD;
  const unsigned short* arow_l = e_lo + (size_t)(rowbase + l16) * kD;
  const int mylab = labels[rowbase + l16];

  float run_m = -1e30f, run_s = 0.0f, coslab = -2.0f;

  // Issue async loads for one 48x128 hi+lo B block (6 x 16B per thread).
  // st2 is the chunk-local stage; global strip = chunk*25 + (st2>>2).
  auto stage_issue = [&](int st2, int bufi) {
    const int j0b = (chunk * 25 + (st2 >> 2)) * 48;   // global column base
    const int k0b = (st2 & 3) * 128;
    const unsigned base_h = lds_addr32(&lsB[bufi][0][0]);
    const unsigned base_l = lds_addr32(&lsB[bufi][1][0]);
#pragma unroll
    for (int it = 0; it < 6; ++it) {
      const int id  = t + it * 256;         // 0..1535
      const int mtx = (id >= 768) ? 1 : 0;
      const int rem = id - mtx * 768;
      const int c   = rem >> 4;             // 0..47
      const int q   = rem & 15;             // 0..15
      const unsigned short* src =
          (mtx ? wt_lo : wt_hi) + (size_t)(j0b + c) * kD + k0b + q * 8;
      const unsigned dst =
          (mtx ? base_l : base_h) + (unsigned)((c * 136 + q * 8) * 2);
      async_g2l_b128(dst, src);
    }
  };

  stage_issue(0, 0);
  int st = 0;
  for (int s = 0; s < 25; ++s) {
    const int strip = chunk * 25 + s;
    const int j0 = strip * 48;
    v8f acc0 = {}, acc1 = {}, acc2 = {};

    for (int kb = 0; kb < 4; ++kb, ++st) {
      wait_asynccnt0();
      __syncthreads();                      // current buffer fully staged
      if (st + 1 < 100) stage_issue(st + 1, (st + 1) & 1);
      const int buf = st & 1;
      const int k0  = kb * 128;

      for (int ks = 0; ks < 4; ++ks) {
        const int kk = ks * 32;
        const v16bf fah = ld_a_frag(arow_h, k0 + kk, akoff);
        const v16bf fal = ld_a_frag(arow_l, k0 + kk, akoff);
        const int bko = kk + ((lane & 16) ? 16 : 0);
#pragma unroll
        for (int tl = 0; tl < 3; ++tl) {
          const unsigned short* pbh = &lsB[buf][0][(tl * 16 + l16) * 136 + bko];
          const unsigned short* pbl = &lsB[buf][1][(tl * 16 + l16) * 136 + bko];
          const v16bf fbh = ld_b_frag(pbh);
          const v16bf fbl = ld_b_frag(pbl);
          v8f& a = (tl == 0 ? acc0 : tl == 1 ? acc1 : acc2);
          a = wmma_bf16(fah, fbh, a);   // hi*hi
          a = wmma_bf16(fah, fbl, a);   // hi*lo
          a = wmma_bf16(fal, fbh, a);   // lo*hi
        }
      }
    }

    // Epilogue: fold column inv-norm, transpose via LDS, 3-group max.
    // (overlaps with the already-issued prefetch of the next strip's block)
#pragma unroll
    for (int tl = 0; tl < 3; ++tl) {
      const float wi = wninv[j0 + tl * 16 + l16];
      const v8f a = (tl == 0 ? acc0 : tl == 1 ? acc1 : acc2);
#pragma unroll
      for (int v = 0; v < 8; ++v)
        lsC[wid][v + hi8][tl * 16 + l16] = a[v] * wi;
    }
    __syncthreads();
#pragma unroll
    for (int i = 0; i < 8; ++i) {
      const int id = lane * 8 + i;       // 0..255 -> (row, class)
      const int r = id >> 4, c = id & 15;
      const float m = fmaxf(fmaxf(lsC[wid][r][3 * c], lsC[wid][r][3 * c + 1]),
                            lsC[wid][r][3 * c + 2]);
      lsC[wid][r][48 + c] = m;
    }
    __syncthreads();

    // Online logsumexp over this strip's 16 classes (lanes 0..15 = rows).
    if (lane < 16) {
      const int cbase = strip * 16;
      for (int c = 0; c < 16; ++c) {
        const float cosv = lsC[wid][l16][48 + c];
        const float lg = kScale * cosv;
        if (cbase + c == mylab) coslab = cosv;
        if (lg > run_m) {
          run_s = run_s * expf(run_m - lg) + 1.0f;
          run_m = lg;
        } else {
          run_s += expf(lg - run_m);
        }
      }
    }
    // next strip's first stage barrier orders lsC reuse
  }

  if (lane < 16) {
    const int row = rowbase + l16;
    out_m[(size_t)chunk * kN + row]  = run_m;
    out_s[(size_t)chunk * kN + row]  = run_s;
    out_cl[(size_t)chunk * kN + row] = coslab;
  }
}

// ---------------------------------------------------------------------------
// 4) Triplet similarity GEMM (e @ e.T) with hi/lo split; per-row hardest
//    positive / negative partials per column chunk.
// ---------------------------------------------------------------------------
__global__ __launch_bounds__(256, 1) void k_sim_gemm(
    const unsigned short* __restrict__ e_hi,
    const unsigned short* __restrict__ e_lo,
    const int* __restrict__ labels,
    float* __restrict__ out_dap, float* __restrict__ out_dan) {
  const int mblk  = blockIdx.x;
  const int chunk = blockIdx.y;
  const int t     = threadIdx.x;
  const int wid   = t >> 5;
  const int lane  = t & 31;
  const int l16   = lane & 15;
  const int hi8   = (lane & 16) ? 8 : 0;
  const int akoff = (lane & 16) ? 16 : 0;
  const int bkoff = (lane & 16) ? 16 : 0;

  const int rowbase = mblk * 128 + wid * 16;
  const unsigned short* arow_h = e_hi + (size_t)(rowbase + l16) * kD;
  const unsigned short* arow_l = e_lo + (size_t)(rowbase + l16) * kD;

  int labr[8];
  float pmax[8], nmin[8];
#pragma unroll
  for (int v = 0; v < 8; ++v) {
    labr[v] = labels[rowbase + hi8 + v];
    pmax[v] = -1.0f;
    nmin[v] = 3.0f;
  }

  const int colbase = chunk * 128;
  for (int ct = 0; ct < 8; ++ct) {
    const int c0 = colbase + ct * 16;
    const unsigned short* brow_h = e_hi + (size_t)(c0 + l16) * kD + bkoff;
    const unsigned short* brow_l = e_lo + (size_t)(c0 + l16) * kD + bkoff;
    v8f acc = {};
    for (int ks = 0; ks < 16; ++ks) {
      const int k0 = ks * 32;
      const v16bf fah = ld_a_frag(arow_h, k0, akoff);
      const v16bf fal = ld_a_frag(arow_l, k0, akoff);
      const v16bf fbh = ld_b_frag(brow_h + k0);
      const v16bf fbl = ld_b_frag(brow_l + k0);
      acc = wmma_bf16(fah, fbh, acc);
      acc = wmma_bf16(fah, fbl, acc);
      acc = wmma_bf16(fal, fbh, acc);
    }
    const int colg = c0 + l16;
    const int labc = labels[colg];
#pragma unroll
    for (int v = 0; v < 8; ++v) {
      const int rowg = rowbase + hi8 + v;
      const float d = sqrtf(fmaxf(2.0f - 2.0f * acc[v], 0.0f));
      const bool same = (labc == labr[v]);
      if (same && (colg != rowg)) pmax[v] = fmaxf(pmax[v], d);
      if (!same)                  nmin[v] = fminf(nmin[v], d);
    }
  }

  // Butterfly reduce across the 16 lanes that share each row.
#pragma unroll
  for (int v = 0; v < 8; ++v) {
    float p = pmax[v], q = nmin[v];
    for (int m = 1; m < 16; m <<= 1) {
      p = fmaxf(p, __shfl_xor(p, m, 32));
      q = fminf(q, __shfl_xor(q, m, 32));
    }
    if (l16 == v) {
      const int rowg = rowbase + hi8 + v;
      out_dap[(size_t)chunk * kN + rowg] = p;
      out_dan[(size_t)chunk * kN + rowg] = q;
    }
  }
}

// ---------------------------------------------------------------------------
// 5) Center loss per row.
// ---------------------------------------------------------------------------
__global__ void k_center(const float* __restrict__ e_n,
                         const float* __restrict__ centers,
                         const int* __restrict__ labels,
                         float* __restrict__ center_row) {
  const int row = blockIdx.x;
  const int t = threadIdx.x;
  __shared__ float red[256];
  const float* c = centers + (size_t)labels[row] * kD;
  const float* e = e_n + (size_t)row * kD;
  const float a0 = e[t] - c[t];
  const float a1 = e[t + 256] - c[t + 256];
  red[t] = a0 * a0 + a1 * a1;
  __syncthreads();
  for (int o = 128; o > 0; o >>= 1) {
    if (t < o) red[t] += red[t + o];
    __syncthreads();
  }
  if (t == 0) center_row[row] = red[0];
}

// ---------------------------------------------------------------------------
// 6) Per-row finalize: merge arc chunk partials (logsumexp + margin
//    correction) and triplet chunk partials.
// ---------------------------------------------------------------------------
__global__ void k_rowfinal(const float* __restrict__ m_arr,
                           const float* __restrict__ s_arr,
                           const float* __restrict__ cl_arr,
                           const float* __restrict__ dap,
                           const float* __restrict__ dan,
                           float* __restrict__ arc_row,
                           float* __restrict__ tl_row) {
  const int n = blockIdx.x * blockDim.x + threadIdx.x;
  if (n >= kN) return;

  float M = -1e30f;
  for (int ch = 0; ch < kArcChunks; ++ch)
    M = fmaxf(M, m_arr[(size_t)ch * kN + n]);
  float S = 0.0f, cl = -2.0f;
  for (int ch = 0; ch < kArcChunks; ++ch) {
    S += s_arr[(size_t)ch * kN + n] * expf(m_arr[(size_t)ch * kN + n] - M);
    cl = fmaxf(cl, cl_arr[(size_t)ch * kN + n]);
  }
  const float lse = M + logf(S);
  const float cc  = fminf(fmaxf(cl, -1.0f + 1e-7f), 1.0f - 1e-7f);
  const float cm  = cosf(acosf(cc) + kMargin);
  const float sc  = kScale * cl;
  const float scm = kScale * cm;
  float corr = 1.0f - expf(sc - lse) + expf(scm - lse);
  corr = fmaxf(corr, 1e-30f);
  arc_row[n] = (lse + logf(corr)) - scm;

  float ap = -1.0f, an = 3.0f;
  for (int ch = 0; ch < kSimChunks; ++ch) {
    ap = fmaxf(ap, dap[(size_t)ch * kN + n]);
    an = fminf(an, dan[(size_t)ch * kN + n]);
  }
  const bool valid = (ap > -0.5f) && (an < 2.5f);
  tl_row[n] = valid ? fmaxf(ap - an + kTripM, 0.0f) : 0.0f;
}

// ---------------------------------------------------------------------------
// 7) Final scalar combine.
// ---------------------------------------------------------------------------
__global__ void k_final(const float* __restrict__ arc_row,
                        const float* __restrict__ tl_row,
                        const float* __restrict__ center_row,
                        float* __restrict__ out) {
  __shared__ float sa[256], st[256], sc[256];
  __shared__ int   sn[256];
  const int t = threadIdx.x;
  float a = 0.0f, tt = 0.0f, cc = 0.0f;
  int nz = 0;
  for (int i = t; i < kN; i += 256) {
    a += arc_row[i];
    const float x = tl_row[i];
    tt += x;
    nz += (x > 0.0f) ? 1 : 0;
    cc += center_row[i];
  }
  sa[t] = a; st[t] = tt; sc[t] = cc; sn[t] = nz;
  __syncthreads();
  for (int o = 128; o > 0; o >>= 1) {
    if (t < o) {
      sa[t] += sa[t + o]; st[t] += st[t + o];
      sc[t] += sc[t + o]; sn[t] += sn[t + o];
    }
    __syncthreads();
  }
  if (t == 0) {
    const float loss_arc  = sa[0] / (float)kN;
    const int   nnz       = sn[0] > 1 ? sn[0] : 1;
    const float loss_trip = st[0] / (float)nnz;
    const float loss_cen  = sc[0] / (float)kN;
    out[0] = loss_arc + 0.5f * loss_trip + kCenterW * loss_cen;
  }
}

// ---------------------------------------------------------------------------
// Host: workspace layout + launch sequence (graph-capture safe).
// ---------------------------------------------------------------------------
extern "C" void kernel_launch(void* const* d_in, const int* in_sizes, int n_in,
                              void* d_out, int out_size, void* d_ws, size_t ws_size,
                              hipStream_t stream) {
  (void)in_sizes; (void)n_in; (void)out_size; (void)ws_size;
  const float* emb     = (const float*)d_in[0];
  const int*   labels  = (const int*)d_in[1];
  const float* W       = (const float*)d_in[2];
  const float* centers = (const float*)d_in[3];

  char* ws = (char*)d_ws;
  size_t off = 0;
  auto alloc = [&](size_t bytes) -> char* {
    char* p = ws + off;
    off += (bytes + 255) & ~(size_t)255;
    return p;
  };

  float*          e_n    = (float*)alloc((size_t)kN * kD * 4);
  unsigned short* e_hi   = (unsigned short*)alloc((size_t)kN * kD * 2);
  unsigned short* e_lo   = (unsigned short*)alloc((size_t)kN * kD * 2);
  unsigned short* wt_hi  = (unsigned short*)alloc((size_t)kCK * kD * 2);
  unsigned short* wt_lo  = (unsigned short*)alloc((size_t)kCK * kD * 2);
  float*          wninv  = (float*)alloc((size_t)kCK * 4);
  float*          arc_m  = (float*)alloc((size_t)kArcChunks * kN * 4);
  float*          arc_s  = (float*)alloc((size_t)kArcChunks * kN * 4);
  float*          arc_cl = (float*)alloc((size_t)kArcChunks * kN * 4);
  float*          dap    = (float*)alloc((size_t)kSimChunks * kN * 4);
  float*          dan    = (float*)alloc((size_t)kSimChunks * kN * 4);
  float*          arcrow = (float*)alloc((size_t)kN * 4);
  float*          tlrow  = (float*)alloc((size_t)kN * 4);
  float*          cenrow = (float*)alloc((size_t)kN * 4);

  k_eprep<<<kN, 256, 0, stream>>>(emb, e_n, e_hi, e_lo);
  k_wprep<<<(kCK + 255) / 256, 256, 0, stream>>>(W, wt_hi, wt_lo, wninv);
  k_arc_gemm<<<dim3(kN / 128, kArcChunks), 256, 0, stream>>>(
      e_hi, e_lo, wt_hi, wt_lo, wninv, labels, arc_m, arc_s, arc_cl);
  k_sim_gemm<<<dim3(kN / 128, kSimChunks), 256, 0, stream>>>(
      e_hi, e_lo, labels, dap, dan);
  k_center<<<kN, 256, 0, stream>>>(e_n, centers, labels, cenrow);
  k_rowfinal<<<(kN + 255) / 256, 256, 0, stream>>>(
      arc_m, arc_s, arc_cl, dap, dan, arcrow, tlrow);
  k_final<<<1, 256, 0, stream>>>(arcrow, tlrow, cenrow, (float*)d_out);
}
